// EGNNLayer_10084583211152
// MI455X (gfx1250) — compile-verified, matrix operand order checked by reading, not run on hardware
//
#include <hip/hip_runtime.h>

typedef __attribute__((ext_vector_type(16))) __bf16 v16bf;
typedef __attribute__((ext_vector_type(8)))  float  v8f;

#define Hdim 128
#define WFRAG_WORDS 256
#define FRAG_WE1 0
#define FRAG_WE2 64
#define FRAG_WC1 96
#define FRAG_WN1 128
#define FRAG_WN2 192
#define N_FRAGS 224
#define SMALL_OFF (N_FRAGS * WFRAG_WORDS)   /* dword offset of small params in ws */
#define SM_BE1 0
#define SM_BE2 128
#define SM_BC1 256
#define SM_BN1 384
#define SM_BN2 512
#define SM_WC2 640
#define SM_WDIST 768
#define AGG_OFF (SMALL_OFF + 896)           /* dword offset of agg buffer in ws */

union FragU { v16bf v; unsigned u[8]; };

// Pack two f32 -> bf16 pair with round-half-up: 2x v_add + 1x v_perm_b32.
static __device__ __forceinline__ unsigned pack_bf16(float a, float b) {
  unsigned ua = __builtin_bit_cast(unsigned, a) + 0x8000u;
  unsigned ub = __builtin_bit_cast(unsigned, b) + 0x8000u;
#if __has_builtin(__builtin_amdgcn_perm)
  // pool = {S0=ub (bytes 4-7), S1=ua (bytes 0-3)}; take ua[3:2], ub[3:2]
  return __builtin_amdgcn_perm(ub, ua, 0x07060302u);
#else
  return (ua >> 16) | (ub & 0xffff0000u);
#endif
}
static __device__ __forceinline__ unsigned short bf16_lo(float v) {
  return (unsigned short)((__builtin_bit_cast(unsigned, v) + 0x8000u) >> 16);
}
static __device__ __forceinline__ float fast_rcp(float x) {
#if __has_builtin(__builtin_amdgcn_rcpf)
  return __builtin_amdgcn_rcpf(x);                   // v_rcp_f32
#else
  return 1.0f / x;
#endif
}
static __device__ __forceinline__ float silu_f(float v) {
  return v * fast_rcp(1.0f + __expf(-v));
}
static __device__ __forceinline__ v16bf load_bfrag(const unsigned* p) {
  const uint4* q = (const uint4*)p;
  uint4 a = q[0];
  uint4 b = q[1];
  FragU f;
  f.u[0]=a.x; f.u[1]=a.y; f.u[2]=a.z; f.u[3]=a.w;
  f.u[4]=b.x; f.u[5]=b.y; f.u[6]=b.z; f.u[7]=b.w;
  return f.v;
}
static __device__ __forceinline__ v8f wmma_bf16(const unsigned a[8], v16bf b, v8f c) {
  FragU fa;
#pragma unroll
  for (int d = 0; d < 8; ++d) fa.u[d] = a[d];
  return __builtin_amdgcn_wmma_f32_16x16x32_bf16(false, fa.v, false, b,
                                                 (short)0, c, false, false);
}

// Build one 16x32 bf16 A-fragment from an f32 row pointer: lane's 2x8
// contiguous k-elements (k = kb + kh*8 + {0..7} and +16) as 2x float4 pairs.
static __device__ __forceinline__ void build_afrag(unsigned a[8], const float* src,
                                                   int kb, int kh) {
  const float* p = src + kb + kh * 8;
#pragma unroll
  for (int half = 0; half < 2; ++half) {
    float4 v0 = *(const float4*)(p + half * 16);
    float4 v1 = *(const float4*)(p + half * 16 + 4);
    a[half*4+0] = pack_bf16(v0.x, v0.y);
    a[half*4+1] = pack_bf16(v0.z, v0.w);
    a[half*4+2] = pack_bf16(v1.x, v1.y);
    a[half*4+3] = pack_bf16(v1.z, v1.w);
  }
}

// Load one 16x32 bf16 A-fragment from the LDS stage (row-major [m][128] bf16):
// per lane 2x 16B contiguous -> 2x ds_load_b128.
static __device__ __forceinline__ void restage_afrag(unsigned a[8],
                                                     const unsigned short* stg,
                                                     int m, int kt, int kh) {
  const uint4 lo = *(const uint4*)&stg[m * Hdim + kt * 32 + kh * 8];
  const uint4 hi = *(const uint4*)&stg[m * Hdim + kt * 32 + 16 + kh * 8];
  a[0]=lo.x; a[1]=lo.y; a[2]=lo.z; a[3]=lo.w;
  a[4]=hi.x; a[5]=hi.y; a[6]=hi.z; a[7]=hi.w;
}

// Software-pipelined K-loop: prefetch B fragment kt+1 while WMMA kt executes.
#define GEMM_KLOOP(NK, AFRAGS, FRAGBASE)                                        \
  {                                                                             \
    v16bf bcur = load_bfrag(wfrag + ((FRAGBASE) + nt) * WFRAG_WORDS + lane * 8);\
    _Pragma("unroll")                                                           \
    for (int kt = 0; kt < (NK); ++kt) {                                         \
      v16bf bnxt = bcur;                                                        \
      if (kt + 1 < (NK))                                                        \
        bnxt = load_bfrag(wfrag + ((FRAGBASE) + (kt + 1) * 8 + nt) *            \
                                      WFRAG_WORDS + lane * 8);                  \
      acc = wmma_bf16((AFRAGS)[kt], bcur, acc);                                 \
      bcur = bnxt;                                                              \
    }                                                                           \
  }

// ---------------------------------------------------------------------------
// Weight pre-pack: f32 (K,128) row-major -> bf16 B-fragments in WMMA lane order
// ---------------------------------------------------------------------------
__global__ __launch_bounds__(256) void egnn_prep_frags(
    const float* __restrict__ We1, const float* __restrict__ We2,
    const float* __restrict__ Wc1, const float* __restrict__ Wn1,
    const float* __restrict__ Wn2, unsigned* __restrict__ wfrag)
{
  int tid = blockIdx.x * 256 + threadIdx.x;
  if (tid >= N_FRAGS * WFRAG_WORDS) return;
  int f = tid >> 8;
  int j = tid & 255;
  int L = j >> 3, d = j & 7;
  int kh = L >> 4;
  const float* W; int g;
  if (f < FRAG_WE2)      { W = We1; g = f; }
  else if (f < FRAG_WC1) { W = We2; g = f - FRAG_WE2; }
  else if (f < FRAG_WN1) { W = Wc1; g = f - FRAG_WC1; }
  else if (f < FRAG_WN2) { W = Wn1; g = f - FRAG_WN1; }
  else                   { W = Wn2; g = f - FRAG_WN2; }
  int kt = g >> 3, nt = g & 7;
  int n = nt * 16 + (L & 15);
  int kloc = (d < 4) ? (kh * 8 + 2 * d) : (16 + kh * 8 + 2 * (d - 4));
  int k = kt * 32 + kloc;
  wfrag[tid] = pack_bf16(W[(long)k * Hdim + n], W[(long)(k + 1) * Hdim + n]);
}

// ---------------------------------------------------------------------------
// Init: zero agg, x_out = x, copy small f32 params into ws
// ---------------------------------------------------------------------------
__global__ __launch_bounds__(256) void egnn_init(
    const float* __restrict__ x,
    const float* __restrict__ be1, const float* __restrict__ be2,
    const float* __restrict__ bc1, const float* __restrict__ bn1,
    const float* __restrict__ bn2, const float* __restrict__ Wc2,
    const float* __restrict__ We1,
    float* __restrict__ small, float* __restrict__ agg,
    float* __restrict__ xout, int N)
{
  long i = (long)blockIdx.x * 256 + threadIdx.x;
  if (i < (long)N * Hdim) agg[i] = 0.0f;
  if (i < (long)N * 3)    xout[i] = x[i];
  if (i < Hdim) {
    int t = (int)i;
    small[SM_BE1 + t] = be1[t];
    small[SM_BE2 + t] = be2[t];
    small[SM_BC1 + t] = bc1[t];
    small[SM_BN1 + t] = bn1[t];
    small[SM_BN2 + t] = bn2[t];
    small[SM_WC2 + t] = Wc2[t];
    small[SM_WDIST + t] = We1[256 * Hdim + t];  // last row of We1 (dist column)
  }
}

// ---------------------------------------------------------------------------
// Edge kernel: one wave per 16-edge tile.
//   l1 = silu([h_row|h_col]@We1[0:256] + dist*We1[256,:] + be1)   (16x128)
//   msg = l1@We2 + be2 ; atomicAdd agg[col] += msg                (16x128)
//   t  = silu(msg@Wc1 + bc1) ; cw = t@Wc2                         (16x1)
//   atomicAdd xout[col] += rel_pos * cw
// ---------------------------------------------------------------------------
__global__ __launch_bounds__(256) void egnn_edge(
    const float* __restrict__ h, const float* __restrict__ x,
    const long long* __restrict__ ei,
    const unsigned* __restrict__ wfrag, const float* __restrict__ small,
    float* __restrict__ agg, float* __restrict__ xout, int N, int E)
{
  __shared__ unsigned short stage[8][16 * Hdim];
  __shared__ float s_dist[8][16];
  __shared__ int   s_col[8][16];
  __shared__ float s_cw[8][16];

  int lane = threadIdx.x & 31;
  int wave = threadIdx.x >> 5;
  long tile = (long)blockIdx.x * 8 + wave;
  long e0 = tile * 16;
  if (e0 >= E) return;
  int m  = lane & 15;
  int kh = lane >> 4;
  long e = e0 + m;
  bool mvalid = (e < (long)E);
  long ec = mvalid ? e : e0;
  int row = (int)ei[ec];
  int col = (int)ei[(long)E + ec];
  float rx = x[(long)row * 3 + 0] - x[(long)col * 3 + 0];
  float ry = x[(long)row * 3 + 1] - x[(long)col * 3 + 1];
  float rz = x[(long)row * 3 + 2] - x[(long)col * 3 + 2];
  float dist = sqrtf(rx * rx + ry * ry + rz * rz);
  if (lane < 16) {
    s_dist[wave][m] = mvalid ? dist : 0.0f;
    s_col[wave][m]  = mvalid ? col : -1;
  }
  float dist_m[8]; int col_m[8];
#pragma unroll
  for (int r = 0; r < 8; ++r) {
    dist_m[r] = s_dist[wave][r + 8 * kh];
    col_m[r]  = s_col[wave][r + 8 * kh];
  }

  // A fragments of [h_row | h_col] (16 x 256 bf16)
  unsigned a1[8][8];
  const float* hr = h + (long)row * Hdim;
  const float* hc = h + (long)col * Hdim;
#pragma unroll
  for (int kt = 0; kt < 8; ++kt)
    build_afrag(a1[kt], (kt < 4) ? hr : hc, (kt & 3) * 32, kh);

  // ---- Layer 1: 8 N-tiles x 8 K-steps of v_wmma_f32_16x16x32_bf16 ----
  for (int nt = 0; nt < 8; ++nt) {
    int n = nt * 16 + m;
    float b  = small[SM_BE1 + n];
    float wd = small[SM_WDIST + n];
    v8f acc;
#pragma unroll
    for (int r = 0; r < 8; ++r) acc[r] = b + dist_m[r] * wd;  // bias + rank-1 dist term
    GEMM_KLOOP(8, a1, FRAG_WE1)
#pragma unroll
    for (int r = 0; r < 8; ++r) {
      int mm = r + 8 * kh;
      stage[wave][mm * Hdim + n] = bf16_lo(silu_f(acc[r]));
    }
  }

  // restage D-layout -> A-layout via LDS (same wave, DS is in-order)
  unsigned a2[4][8];
#pragma unroll
  for (int kt = 0; kt < 4; ++kt)
    restage_afrag(a2[kt], stage[wave], m, kt, kh);

  // ---- Layer 2: edge_msg, fused agg scatter ----
  for (int nt = 0; nt < 8; ++nt) {
    int n = nt * 16 + m;
    float b = small[SM_BE2 + n];
    v8f acc;
#pragma unroll
    for (int r = 0; r < 8; ++r) acc[r] = b;
    GEMM_KLOOP(4, a2, FRAG_WE2)
#pragma unroll
    for (int r = 0; r < 8; ++r) {
      int mm = r + 8 * kh;
      if (col_m[r] >= 0)
        atomicAdd(&agg[(long)col_m[r] * Hdim + n], acc[r]);
      stage[wave][mm * Hdim + n] = bf16_lo(acc[r]);
    }
  }

  unsigned a3[4][8];
#pragma unroll
  for (int kt = 0; kt < 4; ++kt)
    restage_afrag(a3[kt], stage[wave], m, kt, kh);

  // ---- Layer 3: coord gate, fused Wc2 dot via per-lane partials ----
  float cwp[8];
#pragma unroll
  for (int r = 0; r < 8; ++r) cwp[r] = 0.0f;
  for (int nt = 0; nt < 8; ++nt) {
    int n = nt * 16 + m;
    float b  = small[SM_BC1 + n];
    float w2 = small[SM_WC2 + n];
    v8f acc;
#pragma unroll
    for (int r = 0; r < 8; ++r) acc[r] = b;
    GEMM_KLOOP(4, a3, FRAG_WC1)
#pragma unroll
    for (int r = 0; r < 8; ++r) cwp[r] += silu_f(acc[r]) * w2;
  }
  // butterfly sum over the 16 lanes of each half-wave (masks 1,2,4,8 stay in-half)
#pragma unroll
  for (int r = 0; r < 8; ++r) {
    float v = cwp[r];
    v += __shfl_xor(v, 1);
    v += __shfl_xor(v, 2);
    v += __shfl_xor(v, 4);
    v += __shfl_xor(v, 8);
    if (m == 0) s_cw[wave][r + 8 * kh] = v;
  }
  if (lane < 16 && mvalid) {
    float cw = s_cw[wave][m];
    atomicAdd(&xout[(long)col * 3 + 0], rx * cw);
    atomicAdd(&xout[(long)col * 3 + 1], ry * cw);
    atomicAdd(&xout[(long)col * 3 + 2], rz * cw);
  }
}

// ---------------------------------------------------------------------------
// Node kernel: one wave per 16-node tile.
//   h_new = h + silu([h|agg]@Wn1 + bn1)@Wn2 + bn2
// ---------------------------------------------------------------------------
__global__ __launch_bounds__(256) void egnn_node(
    const float* __restrict__ h, const float* __restrict__ agg,
    const unsigned* __restrict__ wfrag, const float* __restrict__ small,
    float* __restrict__ hout, int N)
{
  __shared__ unsigned short stage[8][16 * Hdim];
  int lane = threadIdx.x & 31;
  int wave = threadIdx.x >> 5;
  long tile = (long)blockIdx.x * 8 + wave;
  long n0 = tile * 16;
  if (n0 >= N) return;
  int m  = lane & 15;
  int kh = lane >> 4;
  long node = n0 + m;
  if (node >= N) node = N - 1;

  unsigned a1[8][8];
  const float* hr = h   + node * Hdim;
  const float* ar = agg + node * Hdim;
#pragma unroll
  for (int kt = 0; kt < 8; ++kt)
    build_afrag(a1[kt], (kt < 4) ? hr : ar, (kt & 3) * 32, kh);

  for (int nt = 0; nt < 8; ++nt) {
    int n = nt * 16 + m;
    float b = small[SM_BN1 + n];
    v8f acc;
#pragma unroll
    for (int r = 0; r < 8; ++r) acc[r] = b;
    GEMM_KLOOP(8, a1, FRAG_WN1)
#pragma unroll
    for (int r = 0; r < 8; ++r) {
      int mm = r + 8 * kh;
      stage[wave][mm * Hdim + n] = bf16_lo(silu_f(acc[r]));
    }
  }

  unsigned a2[4][8];
#pragma unroll
  for (int kt = 0; kt < 4; ++kt)
    restage_afrag(a2[kt], stage[wave], m, kt, kh);

  for (int nt = 0; nt < 8; ++nt) {
    int n = nt * 16 + m;
    float b = small[SM_BN2 + n];
    v8f acc;
#pragma unroll
    for (int r = 0; r < 8; ++r) acc[r] = b;
    GEMM_KLOOP(4, a2, FRAG_WN2)
#pragma unroll
    for (int r = 0; r < 8; ++r) {
      long nd = n0 + r + 8 * kh;
      if (nd < N)
        hout[nd * Hdim + n] = h[nd * Hdim + n] + acc[r];
    }
  }
}

// ---------------------------------------------------------------------------
extern "C" void kernel_launch(void* const* d_in, const int* in_sizes, int n_in,
                              void* d_out, int out_size, void* d_ws, size_t ws_size,
                              hipStream_t stream) {
  const float*     h   = (const float*)d_in[0];
  const float*     x   = (const float*)d_in[1];
  const long long* ei  = (const long long*)d_in[2];
  const float*     We1 = (const float*)d_in[3];
  const float*     be1 = (const float*)d_in[4];
  const float*     We2 = (const float*)d_in[5];
  const float*     be2 = (const float*)d_in[6];
  const float*     Wn1 = (const float*)d_in[7];
  const float*     bn1 = (const float*)d_in[8];
  const float*     Wn2 = (const float*)d_in[9];
  const float*     bn2 = (const float*)d_in[10];
  const float*     Wc1 = (const float*)d_in[11];
  const float*     bc1 = (const float*)d_in[12];
  const float*     Wc2 = (const float*)d_in[13];

  int N = in_sizes[0] / Hdim;
  int E = in_sizes[2] / 2;

  float*    ws    = (float*)d_ws;
  unsigned* wfrag = (unsigned*)d_ws;
  float*    small = ws + SMALL_OFF;
  float*    agg   = ws + AGG_OFF;
  float*    hout  = (float*)d_out;
  float*    xout  = hout + (long)N * Hdim;

  // 1) weight pre-pack into bf16 WMMA fragments
  egnn_prep_frags<<<N_FRAGS, 256, 0, stream>>>(We1, We2, Wc1, Wn1, Wn2, wfrag);

  // 2) zero agg, seed x_out, copy small params
  long initN = (long)N * Hdim;
  egnn_init<<<(int)((initN + 255) / 256), 256, 0, stream>>>(
      x, be1, be2, bc1, bn1, bn2, Wc2, We1, small, agg, xout, N);

  // 3) edge MLP + scatters (one wave per 16 edges, 8 waves/block)
  int etiles = (E + 15) / 16;
  egnn_edge<<<(etiles + 7) / 8, 256, 0, stream>>>(
      h, x, ei, wfrag, small, agg, xout, N, E);

  // 4) node MLP + residual
  int ntiles = (N + 15) / 16;
  egnn_node<<<(ntiles + 7) / 8, 256, 0, stream>>>(
      h, agg, wfrag, small, hout, N);
}
